// TGAT_13838384628053
// MI455X (gfx1250) — matrix operand order repaired and checked by reading, not compiled, not run on hardware
//
#include <hip/hip_runtime.h>
#include <cmath>

// ---------------------------------------------------------------------------
// Problem constants (from reference)
// ---------------------------------------------------------------------------
#define BS     2000
#define NROOT  (3 * BS)          // 6000
#define KNBR   10
#define DN     172
#define DE     172
#define DT     100
#define DO     100
#define HEADS  2
#define DH     (DO / HEADS)      // 50
#define MTOT   (NROOT * (1 + KNBR))  // 66000

// padded dims (K-dim multiple of 32 for bf16 WMMA, N padded to 7*16=112)
#define NP     112
#define KP_Q0  288   // 172+100=272 -> 288
#define KP_KV0 448   // 172+172+100=444 -> 448
#define KP_O0  288   // 100+172=272 -> 288
#define KP_Q1  224   // 100+100=200 -> 224
#define KP_KV1 384   // 100+172+100=372 -> 384
#define KP_O1  224   // 100+100=200 -> 224

typedef __attribute__((ext_vector_type(16))) __bf16 v16bf;
typedef __attribute__((ext_vector_type(8)))  __bf16 v8bf;
typedef __attribute__((ext_vector_type(8)))  float  v8f;

// ---------------------------------------------------------------------------
// WMMA fragment helpers (layouts per CDNA5 ISA 7.12.2, wave32)
// ---------------------------------------------------------------------------
// A (16x32 bf16, row-major source with leading dim lda):
//  lane<16 : row M=lane,    K = {k0..k0+7,  k0+16..k0+23}
//  lane>=16: row M=lane-16, K = {k0+8..+15, k0+24..k0+31}
static __device__ inline v16bf load_a_frag(const __bf16* __restrict__ A, int lda,
                                           int m0, int k0, int lane) {
  const int half = (lane >> 4) & 1;
  const int m    = lane & 15;
  const __bf16* row = A + (size_t)(m0 + m) * lda + k0 + half * 8;
  v8bf lo = *(const v8bf*)(row);
  v8bf hi = *(const v8bf*)(row + 16);
  return __builtin_shufflevector(lo, hi, 0,1,2,3,4,5,6,7,8,9,10,11,12,13,14,15);
}

// B (32x16 bf16). Weights stored transposed: Wt[n * ldw + k] (ldw = padded K dim)
//  lane<16 : col N=lane,    K = {k0..k0+7,  k0+16..k0+23}
//  lane>=16: col N=lane-16, K = {k0+8..+15, k0+24..k0+31}
static __device__ inline v16bf load_b_frag(const __bf16* __restrict__ Wt, int ldw,
                                           int n0, int k0, int lane) {
  const int n  = lane & 15;
  const int kb = ((lane >> 4) & 1) * 8;
  const __bf16* col = Wt + (size_t)(n0 + n) * ldw + k0 + kb;
  v8bf lo = *(const v8bf*)(col);
  v8bf hi = *(const v8bf*)(col + 16);
  return __builtin_shufflevector(lo, hi, 0,1,2,3,4,5,6,7,8,9,10,11,12,13,14,15);
}

static __device__ inline v8f wmma_bf16(v16bf a, v16bf b, v8f c) {
  return __builtin_amdgcn_wmma_f32_16x16x32_bf16(false, a, false, b, (short)0, c,
                                                 false, false);
}

// ---------------------------------------------------------------------------
// Weight prep: fp32 [rows x 100] -> bf16 transposed padded [112 x kpad]
// ---------------------------------------------------------------------------
__global__ void convert_wt_kernel(const float* __restrict__ W, __bf16* __restrict__ Wt,
                                  int rows, int kpad) {
  int i = blockIdx.x * 256 + threadIdx.x;
  int total = NP * kpad;
  if (i >= total) return;
  int n = i / kpad;
  int k = i % kpad;
  float v = (k < rows && n < DO) ? W[k * DO + n] : 0.0f;
  Wt[i] = (__bf16)v;
}

// ---------------------------------------------------------------------------
// Fused temporal-attention layer.
// One workgroup (256 threads = 8 waves) handles 16 dst rows (=> 160 kv rows).
// ---------------------------------------------------------------------------
template <int DIN, int DQP, int DKVP, int DOP>
__global__ __launch_bounds__(256, 1)
void attn_kernel(const float* __restrict__ dst_h, int dst_stride,
                 const float* __restrict__ src_h, int src_stride,
                 const float* __restrict__ ef,    // rows (n*K+k) x DE
                 const float* __restrict__ dtv,   // (n*K+k)
                 const __bf16* __restrict__ Wq, const __bf16* __restrict__ Wk,
                 const __bf16* __restrict__ Wv, const __bf16* __restrict__ Wo,
                 const float* __restrict__ bq, const float* __restrict__ bk,
                 const float* __restrict__ bv, const float* __restrict__ bo,
                 const float* __restrict__ tw, const float* __restrict__ tb,
                 const float* __restrict__ gam, const float* __restrict__ bet,
                 float* __restrict__ out, int N) {
  const int n0 = blockIdx.x * 16;
  if (n0 >= N) return;

  const int tid  = threadIdx.x;
  const int wave = tid >> 5;
  const int lane = tid & 31;

  extern __shared__ __align__(16) char smem_raw[];
  __bf16* q_in = (__bf16*)smem_raw;                 // 16 x DQP
  __bf16* kv   = q_in + 16 * DQP;                   // 160 x DKVP
  float*  Qs   = (float*)(kv + 160 * DKVP);         // 16 x 112
  float*  Ks   = Qs + 16 * NP;                      // 160 x 112
  float*  Vs   = Ks + 160 * NP;                     // 160 x 112
  __bf16* oc   = (__bf16*)(Vs + 160 * NP);          // 16 x DOP
  float*  rS   = (float*)(oc + 16 * DOP);           // 16 x 112
  float*  sS   = rS + 16 * NP;                      // 16*K*H = 320 scores

  // ---- Phase 1: build q_in = [dst_h | cos(tb)] (bf16, zero-padded) --------
  for (int i = tid; i < 16 * DQP; i += 256) {
    int r = i / DQP, c = i % DQP;
    float v = 0.0f;
    if (c < DIN)               v = dst_h[(size_t)(n0 + r) * dst_stride + c];
    else if (c < DIN + DT)     v = __cosf(tb[c - DIN]);
    q_in[i] = (__bf16)v;
  }
  // ---- Phase 2: build kv = [src_h | ef | cos(dt*tw+tb)] -------------------
  for (int i = tid; i < 160 * DKVP; i += 256) {
    int rr = i / DKVP, c = i % DKVP;
    int nl = rr / KNBR, k = rr % KNBR;
    size_t srow = (size_t)(n0 + nl) * KNBR + k;
    float v = 0.0f;
    if (c < DIN)                    v = src_h[srow * src_stride + c];
    else if (c < DIN + DE)          v = ef[srow * DE + (c - DIN)];
    else if (c < DIN + DE + DT) {
      int j = c - DIN - DE;
      v = __cosf(dtv[srow] * tw[j] + tb[j]);
    }
    kv[i] = (__bf16)v;
  }
  __syncthreads();

  // ---- Phase 3: WMMA GEMMs (waves 0..6 each own one 16-col tile) ----------
  if (wave < 7) {
    const int n0c  = wave * 16;
    const int ncol = n0c + (lane & 15);
    const int mo   = ((lane >> 4) & 1) * 8;

    // Q = q_in @ Wq  (1 row tile)
    {
      v8f acc = {0.f,0.f,0.f,0.f,0.f,0.f,0.f,0.f};
      for (int k0 = 0; k0 < DQP; k0 += 32) {
        v16bf a = load_a_frag(q_in, DQP, 0, k0, lane);
        v16bf b = load_b_frag(Wq, DQP, n0c, k0, lane);
        acc = wmma_bf16(a, b, acc);
      }
      float bias = (ncol < DO) ? bq[ncol] : 0.0f;
      #pragma unroll
      for (int r = 0; r < 8; r++)
        Qs[(mo + r) * NP + ncol] = (ncol < DO) ? acc[r] + bias : 0.0f;
    }
    // K = kv @ Wk  (10 row tiles, B-fragment reused across row tiles)
    {
      v8f acc[10];
      #pragma unroll
      for (int t = 0; t < 10; t++) acc[t] = (v8f){0.f,0.f,0.f,0.f,0.f,0.f,0.f,0.f};
      for (int k0 = 0; k0 < DKVP; k0 += 32) {
        v16bf b = load_b_frag(Wk, DKVP, n0c, k0, lane);
        #pragma unroll
        for (int rt = 0; rt < 10; rt++) {
          v16bf a = load_a_frag(kv, DKVP, rt * 16, k0, lane);
          acc[rt] = wmma_bf16(a, b, acc[rt]);
        }
      }
      float bias = (ncol < DO) ? bk[ncol] : 0.0f;
      #pragma unroll
      for (int rt = 0; rt < 10; rt++)
        #pragma unroll
        for (int r = 0; r < 8; r++)
          Ks[(rt * 16 + mo + r) * NP + ncol] = (ncol < DO) ? acc[rt][r] + bias : 0.0f;
    }
    // V = kv @ Wv
    {
      v8f acc[10];
      #pragma unroll
      for (int t = 0; t < 10; t++) acc[t] = (v8f){0.f,0.f,0.f,0.f,0.f,0.f,0.f,0.f};
      for (int k0 = 0; k0 < DKVP; k0 += 32) {
        v16bf b = load_b_frag(Wv, DKVP, n0c, k0, lane);
        #pragma unroll
        for (int rt = 0; rt < 10; rt++) {
          v16bf a = load_a_frag(kv, DKVP, rt * 16, k0, lane);
          acc[rt] = wmma_bf16(a, b, acc[rt]);
        }
      }
      float bias = (ncol < DO) ? bv[ncol] : 0.0f;
      #pragma unroll
      for (int rt = 0; rt < 10; rt++)
        #pragma unroll
        for (int r = 0; r < 8; r++)
          Vs[(rt * 16 + mo + r) * NP + ncol] = (ncol < DO) ? acc[rt][r] + bias : 0.0f;
    }
  }
  __syncthreads();

  // ---- Phase 4: attention scores s[n,k,h] = leaky_relu(Q . K, 0.2) --------
  for (int i = tid; i < 16 * KNBR * HEADS; i += 256) {
    int nl = i / (KNBR * HEADS);
    int rem = i % (KNBR * HEADS);
    int k = rem / HEADS, h = rem % HEADS;
    float s = 0.0f;
    const float* qrow = Qs + nl * NP + h * DH;
    const float* krow = Ks + (nl * KNBR + k) * NP + h * DH;
    for (int d = 0; d < DH; d++) s += qrow[d] * krow[d];
    s = (s >= 0.0f) ? s : 0.2f * s;
    sS[(nl * KNBR + k) * HEADS + h] = s;
  }
  __syncthreads();

  // ---- Phase 5: softmax over K=10 per (n,h) -------------------------------
  if (tid < 16 * HEADS) {
    int nl = tid / HEADS, h = tid % HEADS;
    float mx = -1e30f;
    for (int k = 0; k < KNBR; k++) mx = fmaxf(mx, sS[(nl * KNBR + k) * HEADS + h]);
    float sum = 0.0f;
    float e[KNBR];
    for (int k = 0; k < KNBR; k++) {
      e[k] = __expf(sS[(nl * KNBR + k) * HEADS + h] - mx);
      sum += e[k];
    }
    float inv = 1.0f / sum;
    for (int k = 0; k < KNBR; k++) sS[(nl * KNBR + k) * HEADS + h] = e[k] * inv;
  }
  __syncthreads();

  // ---- Phase 6: o = sum_k a*V ; oc = [o | dst_h] (bf16) -------------------
  for (int i = tid; i < 16 * DOP; i += 256) {
    int r = i / DOP, c = i % DOP;
    float v = 0.0f;
    if (c < DO) {
      int h = c / DH;
      for (int k = 0; k < KNBR; k++)
        v += sS[(r * KNBR + k) * HEADS + h] * Vs[(r * KNBR + k) * NP + c];
    } else if (c < DO + DIN) {
      v = dst_h[(size_t)(n0 + r) * dst_stride + (c - DO)];
    }
    oc[i] = (__bf16)v;
  }
  __syncthreads();

  // ---- Phase 7: r = relu(oc @ Wo + bo) ------------------------------------
  if (wave < 7) {
    const int n0c  = wave * 16;
    const int ncol = n0c + (lane & 15);
    const int mo   = ((lane >> 4) & 1) * 8;
    v8f acc = {0.f,0.f,0.f,0.f,0.f,0.f,0.f,0.f};
    for (int k0 = 0; k0 < DOP; k0 += 32) {
      v16bf a = load_a_frag(oc, DOP, 0, k0, lane);
      v16bf b = load_b_frag(Wo, DOP, n0c, k0, lane);
      acc = wmma_bf16(a, b, acc);
    }
    float bias = (ncol < DO) ? bo[ncol] : 0.0f;
    #pragma unroll
    for (int r = 0; r < 8; r++)
      rS[(mo + r) * NP + ncol] = (ncol < DO) ? fmaxf(acc[r] + bias, 0.0f) : 0.0f;
  }
  __syncthreads();

  // ---- Phase 8: layernorm + affine, write out -----------------------------
  if (tid < 16) {
    const float* row = rS + tid * NP;
    float mu = 0.0f;
    for (int c = 0; c < DO; c++) mu += row[c];
    mu *= (1.0f / DO);
    float var = 0.0f;
    for (int c = 0; c < DO; c++) { float d = row[c] - mu; var += d * d; }
    var *= (1.0f / DO);
    float inv = rsqrtf(var + 1e-5f);
    float* orow = out + (size_t)(n0 + tid) * DO;
    for (int c = 0; c < DO; c++)
      orow[c] = (row[c] - mu) * inv * gam[c] + bet[c];
  }
}

// ---------------------------------------------------------------------------
// Edge predictor: out[which*BS+i] = relu(src@Ws + dst@Wd + bs + bd) @ Wout + bout
// ---------------------------------------------------------------------------
__global__ __launch_bounds__(128)
void edge_pred_kernel(const float* __restrict__ emb,
                      const float* __restrict__ Ws, const float* __restrict__ bs,
                      const float* __restrict__ Wd, const float* __restrict__ bd,
                      const float* __restrict__ Wout, const float* __restrict__ bout,
                      float* __restrict__ out) {
  int b = blockIdx.x;
  int which = b / BS;       // 0: (src,dst)  1: (src,neg)
  int i = b % BS;
  const float* s = emb + (size_t)i * DO;
  const float* d = emb + (size_t)((which ? 2 * BS : BS) + i) * DO;
  __shared__ float sh_s[DO], sh_d[DO], red[128];
  int tid = threadIdx.x;
  for (int c = tid; c < DO; c += 128) { sh_s[c] = s[c]; sh_d[c] = d[c]; }
  __syncthreads();
  float part = 0.0f;
  for (int c = tid; c < DO; c += 128) {
    float h = bs[c] + bd[c];
    for (int t = 0; t < DO; t++)
      h += sh_s[t] * Ws[t * DO + c] + sh_d[t] * Wd[t * DO + c];
    h = fmaxf(h, 0.0f);
    part += h * Wout[c];
  }
  red[tid] = part;
  __syncthreads();
  for (int st = 64; st > 0; st >>= 1) {
    if (tid < st) red[tid] += red[tid + st];
    __syncthreads();
  }
  if (tid == 0) out[which * BS + i] = red[0] + bout[0];
}

// ---------------------------------------------------------------------------
// Host side
// ---------------------------------------------------------------------------
static inline size_t smem_bytes(int DQP, int DKVP, int DOP) {
  return (size_t)2 * 16 * DQP + (size_t)2 * 160 * DKVP +
         (size_t)4 * 16 * NP + (size_t)4 * 160 * NP * 2 +
         (size_t)2 * 16 * DOP + (size_t)4 * 16 * NP + (size_t)4 * 320;
}

extern "C" void kernel_launch(void* const* d_in, const int* in_sizes, int n_in,
                              void* d_out, int out_size, void* d_ws, size_t ws_size,
                              hipStream_t stream) {
  // input map (setup_inputs insertion order, dicts flattened in insertion order)
  const float* f1_dst = (const float*)d_in[0];
  const float* f2_src = (const float*)d_in[1];
  const float* e2     = (const float*)d_in[2];
  const float* dt2    = (const float*)d_in[3];
  const float* e1     = (const float*)d_in[4];
  const float* dt1    = (const float*)d_in[5];
  // params0
  const float* Wq0 = (const float*)d_in[6];  const float* bq0 = (const float*)d_in[7];
  const float* Wk0 = (const float*)d_in[8];  const float* bk0 = (const float*)d_in[9];
  const float* Wv0 = (const float*)d_in[10]; const float* bv0 = (const float*)d_in[11];
  const float* Wo0 = (const float*)d_in[12]; const float* bo0 = (const float*)d_in[13];
  const float* tw0 = (const float*)d_in[14]; const float* tb0 = (const float*)d_in[15];
  const float* g0  = (const float*)d_in[16]; const float* b0  = (const float*)d_in[17];
  // params1
  const float* Wq1 = (const float*)d_in[18]; const float* bq1 = (const float*)d_in[19];
  const float* Wk1 = (const float*)d_in[20]; const float* bk1 = (const float*)d_in[21];
  const float* Wv1 = (const float*)d_in[22]; const float* bv1 = (const float*)d_in[23];
  const float* Wo1 = (const float*)d_in[24]; const float* bo1 = (const float*)d_in[25];
  const float* tw1 = (const float*)d_in[26]; const float* tb1 = (const float*)d_in[27];
  const float* g1  = (const float*)d_in[28]; const float* b1  = (const float*)d_in[29];
  // pred
  const float* Ws   = (const float*)d_in[30]; const float* bs   = (const float*)d_in[31];
  const float* Wd   = (const float*)d_in[32]; const float* bd   = (const float*)d_in[33];
  const float* Wout = (const float*)d_in[34]; const float* bout = (const float*)d_in[35];

  // workspace layout
  __bf16* wsb = (__bf16*)d_ws;
  size_t off = 0;
  __bf16* wq0 = wsb + off; off += (size_t)KP_Q0  * NP;
  __bf16* wk0 = wsb + off; off += (size_t)KP_KV0 * NP;
  __bf16* wv0 = wsb + off; off += (size_t)KP_KV0 * NP;
  __bf16* wo0 = wsb + off; off += (size_t)KP_O0  * NP;
  __bf16* wq1 = wsb + off; off += (size_t)KP_Q1  * NP;
  __bf16* wk1 = wsb + off; off += (size_t)KP_KV1 * NP;
  __bf16* wv1 = wsb + off; off += (size_t)KP_KV1 * NP;
  __bf16* wo1 = wsb + off; off += (size_t)KP_O1  * NP;
  float* h1  = (float*)((char*)d_ws + ((off * 2 + 15) & ~(size_t)15));
  float* emb = h1 + (size_t)MTOT * DO;

  // weight prep (tiny)
  auto conv = [&](const float* W, __bf16* Wt, int rows, int kpad) {
    int total = NP * kpad;
    convert_wt_kernel<<<(total + 255) / 256, 256, 0, stream>>>(W, Wt, rows, kpad);
  };
  conv(Wq0, wq0, DN + DT,      KP_Q0);
  conv(Wk0, wk0, DN + DE + DT, KP_KV0);
  conv(Wv0, wv0, DN + DE + DT, KP_KV0);
  conv(Wo0, wo0, DO + DN,      KP_O0);
  conv(Wq1, wq1, DO + DT,      KP_Q1);
  conv(Wk1, wk1, DO + DE + DT, KP_KV1);
  conv(Wv1, wv1, DO + DE + DT, KP_KV1);
  conv(Wo1, wo1, DO + DO,      KP_O1);

  const size_t smem0 = smem_bytes(KP_Q0, KP_KV0, KP_O0);
  const size_t smem1 = smem_bytes(KP_Q1, KP_KV1, KP_O1);
  hipFuncSetAttribute(reinterpret_cast<const void*>(&attn_kernel<DN, KP_Q0, KP_KV0, KP_O0>),
                      hipFuncAttributeMaxDynamicSharedMemorySize, (int)smem0);
  hipFuncSetAttribute(reinterpret_cast<const void*>(&attn_kernel<DO, KP_Q1, KP_KV1, KP_O1>),
                      hipFuncAttributeMaxDynamicSharedMemorySize, (int)smem1);

  // layer 0: N = M = 66000 (exactly 4125 blocks of 16 rows)
  attn_kernel<DN, KP_Q0, KP_KV0, KP_O0><<<MTOT / 16, 256, smem0, stream>>>(
      f1_dst, DN, f2_src, DN, e2, dt2,
      wq0, wk0, wv0, wo0, bq0, bk0, bv0, bo0, tw0, tb0, g0, b0,
      h1, MTOT);

  // layer 1: N = NROOT = 6000 (375 blocks); src rows = h1[NROOT + n*K + k]
  attn_kernel<DO, KP_Q1, KP_KV1, KP_O1><<<NROOT / 16, 256, smem1, stream>>>(
      h1, DO, h1 + (size_t)NROOT * DO, DO, e1, dt1,
      wq1, wk1, wv1, wo1, bq1, bk1, bv1, bo1, tw1, tb1, g1, b1,
      emb, NROOT);

  // edge predictor: 2*BS outputs
  edge_pred_kernel<<<2 * BS, 128, 0, stream>>>(emb, Ws, bs, Wd, bd, Wout, bout,
                                               (float*)d_out);
}
